// LayerStacks_24721831756095
// MI455X (gfx1250) — matrix operand order repaired and checked by reading, not compiled, not run on hardware
//
#include <hip/hip_runtime.h>

#define IN_W   1024   // 2*L1
#define LX     32
#define NBUCK  8

typedef __attribute__((ext_vector_type(16))) _Float16 v16h;
typedef __attribute__((ext_vector_type(8)))  _Float16 v8h;
typedef __attribute__((ext_vector_type(8)))  float    v8f;
typedef __attribute__((ext_vector_type(4)))  float    v4f;

__device__ __forceinline__ float clamp01(float v) {
    return __builtin_fminf(__builtin_fmaxf(v, 0.0f), 1.0f);
}

// ---- meta layout (64 ints): [0..7]=counts, [8..15]=fill, [16..24]=bucketStart, [32]=numTiles
__global__ void k_init(int* meta) {
    if (threadIdx.x < 64) meta[threadIdx.x] = 0;
}

__global__ void k_count(const int* __restrict__ lsidx, int B, int* __restrict__ meta) {
    int i = blockIdx.x * blockDim.x + threadIdx.x;
    if (i < B) atomicAdd(&meta[lsidx[i] & 7], 1);
}

// 32-row tiles, each entirely inside one bucket
__global__ void k_build(int* __restrict__ meta, int* __restrict__ tiles) {
    if (threadIdx.x != 0) return;
    int s = 0;
    for (int k = 0; k < NBUCK; k++) { meta[16 + k] = s; s += meta[k]; }
    meta[24] = s;
    int nt = 0;
    for (int k = 0; k < NBUCK; k++) {
        int cnt = meta[k], st = meta[16 + k];
        for (int t = 0; t < cnt; t += 32) {
            tiles[2 * nt] = k;
            tiles[2 * nt + 1] = st + t;
            nt++;
        }
    }
    meta[32] = nt;
}

__global__ void k_scatter(const int* __restrict__ lsidx, int B,
                          int* __restrict__ meta, int* __restrict__ rowIdx) {
    int i = blockIdx.x * blockDim.x + threadIdx.x;
    if (i < B) {
        int k = lsidx[i] & 7;
        int pos = meta[16 + k] + atomicAdd(&meta[8 + k], 1);
        rowIdx[pos] = i;
    }
}

// ---- pre-swizzle (w0 + w_fact) into per-lane WMMA B-fragments, f16.
// frag index = ((bucket*2 + ntile)*32 + kstep)*512 + lane*16 + e
// Kloc = 8*(lane>>4)+e for e<8, 16+8*(lane>>4)+(e-8) for e>=8; N = ntile*16 + (lane&15)
__global__ void k_prep_w0(const float* __restrict__ w0, const float* __restrict__ wfact,
                          _Float16* __restrict__ frag) {
    int gid = blockIdx.x * blockDim.x + threadIdx.x;
    if (gid >= NBUCK * 2 * 32 * 32 * 16) return;
    int e      = gid & 15;
    int lane   = (gid >> 4) & 31;
    int ks     = (gid >> 9) & 31;
    int nt     = (gid >> 14) & 1;
    int bucket = gid >> 15;
    int Kg = ks * 32 + ((e < 8) ? ((lane >> 4) * 8 + e) : (16 + (lane >> 4) * 8 + (e - 8)));
    int N  = nt * 16 + (lane & 15);
    float v = w0[(size_t)(bucket * 32 + N) * IN_W + Kg] + wfact[(size_t)N * IN_W + Kg];
    frag[gid] = (_Float16)v;
}

__global__ void k_prep_w1(const float* __restrict__ w1, _Float16* __restrict__ frag) {
    int gid = blockIdx.x * blockDim.x + threadIdx.x;
    if (gid >= NBUCK * 2 * 32 * 16) return;
    int e      = gid & 15;
    int lane   = (gid >> 4) & 31;
    int nt     = (gid >> 9) & 1;
    int bucket = gid >> 10;
    int Kloc = (e < 8) ? ((lane >> 4) * 8 + e) : (16 + (lane >> 4) * 8 + (e - 8));
    int N    = nt * 16 + (lane & 15);
    frag[gid] = (_Float16)w1[(bucket * 32 + N) * LX + Kloc];
}

// Build an f16 A-fragment from 4 aligned float4 chunks of one x row
__device__ __forceinline__ v16h load_a_frag(const float* pa) {
    v4f fa = *(const v4f*)(pa);
    v4f fb = *(const v4f*)(pa + 4);
    v4f fc = *(const v4f*)(pa + 16);
    v4f fd = *(const v4f*)(pa + 20);
    v16h a;
    #pragma unroll
    for (int i = 0; i < 4; i++) {
        a[i]      = (_Float16)fa[i];
        a[4 + i]  = (_Float16)fb[i];
        a[8 + i]  = (_Float16)fc[i];
        a[12 + i] = (_Float16)fd[i];
    }
    return a;
}

// ---- main: one wave = one 32-row tile (two 16-row M-tiles) of a single bucket.
// Both M-tiles share the same B fragments -> 4 WMMAs per 2 B-loads per K-step.
__global__ void __launch_bounds__(32)
k_main(const float* __restrict__ x,
       const float* __restrict__ b0, const float* __restrict__ b1,
       const float* __restrict__ w2, const float* __restrict__ b2,
       const int* __restrict__ meta, const int* __restrict__ tiles,
       const int* __restrict__ rowIdx,
       const _Float16* __restrict__ w0frag, const _Float16* __restrict__ w1frag,
       float* __restrict__ out) {
    __shared__ __align__(16) _Float16 h0[32 * 32];

    int tile = blockIdx.x;
    if (tile >= meta[32]) return;          // wave-uniform: EXEC stays all-ones
    int bucket    = tiles[2 * tile];
    int listStart = tiles[2 * tile + 1];
    int bucketEnd = meta[16 + bucket] + meta[bucket];

    int lane = threadIdx.x;
    int mrow = lane & 15;                  // matrix row this lane feeds (A layout)
    int ncol = lane & 15;                  // matrix col this lane holds (C/D layout)
    int koff = (lane < 16) ? 0 : 8;
    int rbase = (lane < 16) ? 0 : 8;

    int lpA = listStart + mrow;
    int lpB = listStart + 16 + mrow;
    int rowA = rowIdx[(lpA < bucketEnd) ? lpA : listStart];   // dummy=first row for tail
    int rowB = rowIdx[(lpB < bucketEnd) ? lpB : listStart];
    const float* xa = x + (size_t)rowA * IN_W;
    const float* xb = x + (size_t)rowB * IN_W;

    // layer-0 accumulators seeded with bias (broadcast over M)
    v8f c0a, c1a, c0b, c1b;
    {
        float bias0 = b0[bucket * 32 + ncol];
        float bias1 = b0[bucket * 32 + 16 + ncol];
        #pragma unroll
        for (int r = 0; r < 8; r++) {
            c0a[r] = bias0; c1a[r] = bias1;
            c0b[r] = bias0; c1b[r] = bias1;
        }
    }
    const _Float16* f0base = w0frag + (size_t)(bucket * 2 + 0) * 32 * 512 + lane * 16;
    const _Float16* f1base = w0frag + (size_t)(bucket * 2 + 1) * 32 * 512 + lane * 16;

    #pragma unroll 2
    for (int ks = 0; ks < 32; ks++) {
        const float* pa = xa + ks * 32 + koff;
        const float* pb = xb + ks * 32 + koff;
        // speculative SYS-scope prefetch (dropped silently past buffer end)
        __builtin_prefetch(pa + 160, 0, 0);
        __builtin_prefetch(pb + 160, 0, 0);
        v16h aA = load_a_frag(pa);
        v16h aB = load_a_frag(pb);
        v16h bf0 = *(const v16h*)(f0base + ks * 512);
        v16h bf1 = *(const v16h*)(f1base + ks * 512);
        c0a = __builtin_amdgcn_wmma_f32_16x16x32_f16(false, aA, false, bf0, (short)0, c0a, false, false);
        c1a = __builtin_amdgcn_wmma_f32_16x16x32_f16(false, aA, false, bf1, (short)0, c1a, false, false);
        c0b = __builtin_amdgcn_wmma_f32_16x16x32_f16(false, aB, false, bf0, (short)0, c0b, false, false);
        c1b = __builtin_amdgcn_wmma_f32_16x16x32_f16(false, aB, false, bf1, (short)0, c1b, false, false);
    }

    // clamp -> LDS (f16 row-major [32][32]) for the cross-lane transpose
    #pragma unroll
    for (int r = 0; r < 8; r++) {
        h0[(rbase + r) * 32 + ncol]           = (_Float16)clamp01(c0a[r]);
        h0[(rbase + r) * 32 + 16 + ncol]      = (_Float16)clamp01(c1a[r]);
        h0[(16 + rbase + r) * 32 + ncol]      = (_Float16)clamp01(c0b[r]);
        h0[(16 + rbase + r) * 32 + 16 + ncol] = (_Float16)clamp01(c1b[r]);
    }
    __syncthreads();

    // layer 1: A from LDS, B pre-swizzled (shared across both M-tiles)
    v16h a1A, a1B;
    {
        v8h lo = *(const v8h*)(h0 + mrow * 32 + koff);
        v8h hi = *(const v8h*)(h0 + mrow * 32 + 16 + koff);
        v8h lo2 = *(const v8h*)(h0 + (16 + mrow) * 32 + koff);
        v8h hi2 = *(const v8h*)(h0 + (16 + mrow) * 32 + 16 + koff);
        #pragma unroll
        for (int i = 0; i < 8; i++) {
            a1A[i] = lo[i];  a1A[8 + i] = hi[i];
            a1B[i] = lo2[i]; a1B[8 + i] = hi2[i];
        }
    }
    v8f d0a, d1a, d0b, d1b;
    {
        float bb0 = b1[bucket * 32 + ncol];
        float bb1 = b1[bucket * 32 + 16 + ncol];
        #pragma unroll
        for (int r = 0; r < 8; r++) {
            d0a[r] = bb0; d1a[r] = bb1;
            d0b[r] = bb0; d1b[r] = bb1;
        }
    }
    v16h g0 = *(const v16h*)(w1frag + (size_t)(bucket * 2 + 0) * 512 + lane * 16);
    v16h g1 = *(const v16h*)(w1frag + (size_t)(bucket * 2 + 1) * 512 + lane * 16);
    d0a = __builtin_amdgcn_wmma_f32_16x16x32_f16(false, a1A, false, g0, (short)0, d0a, false, false);
    d1a = __builtin_amdgcn_wmma_f32_16x16x32_f16(false, a1A, false, g1, (short)0, d1a, false, false);
    d0b = __builtin_amdgcn_wmma_f32_16x16x32_f16(false, a1B, false, g0, (short)0, d0b, false, false);
    d1b = __builtin_amdgcn_wmma_f32_16x16x32_f16(false, a1B, false, g1, (short)0, d1b, false, false);

    // layer 2: 32-long dot per row via VALU + half-wave shuffle reduction
    float wA = w2[bucket * 32 + ncol];
    float wB = w2[bucket * 32 + 16 + ncol];
    float bias2 = b2[bucket];
    #pragma unroll
    for (int r = 0; r < 8; r++) {
        float pA = clamp01(d0a[r]) * wA + clamp01(d1a[r]) * wB;
        float pB = clamp01(d0b[r]) * wA + clamp01(d1b[r]) * wB;
        pA += __shfl_xor(pA, 1, 32);
        pB += __shfl_xor(pB, 1, 32);
        pA += __shfl_xor(pA, 2, 32);
        pB += __shfl_xor(pB, 2, 32);
        pA += __shfl_xor(pA, 4, 32);
        pB += __shfl_xor(pB, 4, 32);
        pA += __shfl_xor(pA, 8, 32);
        pB += __shfl_xor(pB, 8, 32);
        if ((lane & 15) == 0) {
            int lp = listStart + rbase + r;
            if (lp < bucketEnd) out[rowIdx[lp]] = pA + bias2;
            int lp2 = listStart + 16 + rbase + r;
            if (lp2 < bucketEnd) out[rowIdx[lp2]] = pB + bias2;
        }
    }
}

extern "C" void kernel_launch(void* const* d_in, const int* in_sizes, int n_in,
                              void* d_out, int out_size, void* d_ws, size_t ws_size,
                              hipStream_t stream) {
    const float* x     = (const float*)d_in[0];
    const int*   lsidx = (const int*)  d_in[1];
    const float* wfact = (const float*)d_in[2];
    const float* w0    = (const float*)d_in[3];
    const float* b0    = (const float*)d_in[4];
    const float* w1    = (const float*)d_in[5];
    const float* b1    = (const float*)d_in[6];
    const float* w2    = (const float*)d_in[7];
    const float* b2    = (const float*)d_in[8];
    float* out = (float*)d_out;

    int B = in_sizes[0] / IN_W;
    int maxTiles = (B + 31) / 32 + NBUCK;

    char* ws = (char*)d_ws;
    int* meta  = (int*)ws;                                     // 256 B
    int* tiles = (int*)(ws + 256);                             // 2*maxTiles ints
    size_t off = 256 + (((size_t)2 * maxTiles * sizeof(int) + 255) & ~(size_t)255);
    int* rowIdx = (int*)(ws + off);
    off += (((size_t)B * sizeof(int) + 255) & ~(size_t)255);
    _Float16* w0frag = (_Float16*)(ws + off);
    off += (size_t)NBUCK * 2 * 32 * 512 * sizeof(_Float16);    // 512 KB
    _Float16* w1frag = (_Float16*)(ws + off);
    (void)ws_size; (void)n_in; (void)out_size;

    int nb = (B + 255) / 256;
    k_init   <<<1, 64, 0, stream>>>(meta);
    k_count  <<<nb, 256, 0, stream>>>(lsidx, B, meta);
    k_build  <<<1, 32, 0, stream>>>(meta, tiles);
    k_scatter<<<nb, 256, 0, stream>>>(lsidx, B, meta, rowIdx);
    k_prep_w0<<<(NBUCK * 2 * 32 * 32 * 16 + 255) / 256, 256, 0, stream>>>(w0, wfact, w0frag);
    k_prep_w1<<<(NBUCK * 2 * 32 * 16 + 255) / 256, 256, 0, stream>>>(w1, w1frag);
    k_main   <<<maxTiles, 32, 0, stream>>>(x, b0, b1, w2, b2, meta, tiles, rowIdx,
                                           w0frag, w1frag, out);
}
